// CBAM_71262097375889
// MI455X (gfx1250) — matrix operand order repaired and checked by reading, not compiled, not run on hardware
//
#include <hip/hip_runtime.h>
#include <hip/hip_bf16.h>
#include <math.h>
#include <stdint.h>

typedef __attribute__((ext_vector_type(16))) _Float16 v16h;
typedef __attribute__((ext_vector_type(8)))  float    v8f;
typedef __attribute__((ext_vector_type(4)))  unsigned int v4u;
typedef __attribute__((ext_vector_type(8)))  int      v8i;
typedef __attribute__((ext_vector_type(4)))  int      v4i;

#define Bq 4
#define Cc 32
#define Hh 128
#define Ww 128
#define HW (Hh * Ww)
#define KK9 9
#define MH 4
#define MS 512
#define HD 8

union Frag16 { v16h vec; _Float16 e[16]; };

__device__ __forceinline__ float sigm(float v) { return 1.0f / (1.0f + __expf(-v)); }

// -------------------------------------------------------------- TDM helper
// 1-D contiguous DMA of nElem f16 values from global to LDS via the Tensor
// Data Mover.  D# group0: count=1, lds_addr, 57-bit global addr, type=2.
// group1: data_size=1 (2B), tensor_dim0 = tile_dim0 = nElem, stride = nElem.
__device__ __forceinline__ void tdm_load_f16(_Float16* ldsPtr,
                                             const _Float16* gPtr,
                                             unsigned int nElem) {
  const uint64_t ga = (uint64_t)(uintptr_t)gPtr;
  const unsigned int ldsOff = (unsigned int)(uintptr_t)ldsPtr;  // low 32 = LDS offset
  v4u g0;
  g0[0] = 1u;                                                // count = 1
  g0[1] = ldsOff;                                            // lds_addr
  g0[2] = (unsigned int)ga;                                  // global_addr[31:0]
  g0[3] = (unsigned int)((ga >> 32) & 0x01FFFFFFu) | (2u << 30);  // addr[56:32] | type=2
  v8i g1;
  g1[0] = (int)(1u << 16);                                   // data_size=1 (2B), mask=0
  g1[1] = (int)((nElem & 0xFFFFu) << 16);                    // tensor_dim0[15:0]
  g1[2] = (int)((nElem >> 16) & 0xFFFFu);                    // tensor_dim0[31:16]
  g1[3] = (int)((nElem & 0xFFFFu) << 16);                    // tile_dim0
  g1[4] = 0;                                                 // tile_dim1/2 unused
  g1[5] = (int)nElem;                                        // tensor_dim0_stride[31:0]
  g1[6] = 0;
  g1[7] = 0;
  v4i z4 = {0, 0, 0, 0};
#if defined(__clang_major__) && __clang_major__ >= 23
  v8i z8 = {0, 0, 0, 0, 0, 0, 0, 0};
  __builtin_amdgcn_tensor_load_to_lds(g0, g1, z4, z4, z8, 0);
#else
  __builtin_amdgcn_tensor_load_to_lds(g0, g1, z4, z4, 0);
#endif
}

// ---------------------------------------------------------------- kernel 0
// Precompute f16 weight tables + pre-swizzled WMMA B-fragment tables for the
// attention memory (content depends only on (hd, mt, lane, slot)).
__global__ __launch_bounds__(256) void cbam_prep(const float* __restrict__ dcn_w,
                                                 const float* __restrict__ sw_w1,
                                                 const float* __restrict__ memw,
                                                 _Float16* __restrict__ wTab,
                                                 _Float16* __restrict__ w1Tab,
                                                 _Float16* __restrict__ scoreB,
                                                 _Float16* __restrict__ recB) {
  int i = blockIdx.x * 256 + threadIdx.x;
  if (i < KK9 * Cc * Cc) {                       // dcn_w -> [k][m][ic] f16
    const int k = i / (Cc * Cc);
    const int r = i % (Cc * Cc);
    const int m = r / Cc, ic = r % Cc;
    wTab[i] = (_Float16)dcn_w[(m * Cc + ic) * KK9 + k];
    return;
  }
  i -= KK9 * Cc * Cc;
  if (i < Cc * 64) {                             // sw_w1 -> f16 [o][c]
    w1Tab[i] = (_Float16)sw_w1[i];
    return;
  }
  i -= Cc * 64;
  if (i < MH * 32 * 32 * 16) {                   // score B-fragment table
    const int hh = i & 15, lane = (i >> 4) & 31, mt = (i >> 9) & 31, hd = i >> 14;
    const int n = lane & 15, hf = lane >> 4;
    const int kk = hf * 16 + hh, d = kk - hd * HD;
    scoreB[i] = (_Float16)((d >= 0 && d < HD)
                               ? memw[(size_t)(hd * MS + mt * 16 + n) * HD + d]
                               : 0.f);
    return;
  }
  i -= MH * 32 * 32 * 16;
  if (i < MH * 16 * 32 * 16) {                   // rec B-fragment table
    const int hh = i & 15, lane = (i >> 4) & 31, mt = (i >> 9) & 15, hd = i >> 13;
    const int n = lane & 15, hf = lane >> 4;
    const int kk = hf * 16 + hh;
    recB[i] = (_Float16)((n < HD)
                             ? memw[(size_t)(hd * MS + mt * 32 + kk) * HD + n]
                             : 0.f);
  }
}

// ---------------------------------------------------------------- kernel 1
__global__ __launch_bounds__(256) void cbam_mean(const float* __restrict__ x,
                                                 float* __restrict__ yAvg) {
  __shared__ float red[256];
  const int bc = blockIdx.x;
  const float* p = x + (size_t)bc * HW;
  float s = 0.f;
  for (int i = threadIdx.x; i < HW; i += 256) s += p[i];
  red[threadIdx.x] = s;
  __syncthreads();
  for (int o = 128; o > 0; o >>= 1) {
    if (threadIdx.x < o) red[threadIdx.x] += red[threadIdx.x + o];
    __syncthreads();
  }
  if (threadIdx.x == 0) yAvg[bc] = red[0] * (1.0f / (float)HW);
}

// ---------------------------------------------------------------- kernel 2
__global__ __launch_bounds__(128) void cbam_se(const float* __restrict__ yAvg,
                                               const float* __restrict__ fs_w1,
                                               const float* __restrict__ fs_w2,
                                               const float* __restrict__ fc_w1,
                                               const float* __restrict__ fc_w2,
                                               float* __restrict__ ySp,
                                               float* __restrict__ yCh) {
  const int t = threadIdx.x;
  if (t >= Bq * Cc) return;
  const int b = t >> 5, c = t & 31;
  const float* ya = yAvg + b * Cc;
  float o1 = 0.f;
  for (int j = 0; j < 2; ++j) {
    float hsum = 0.f;
    for (int k = 0; k < Cc; ++k) hsum += ya[k] * fs_w1[j * Cc + k];
    o1 += fmaxf(hsum, 0.f) * fs_w2[c * 2 + j];
  }
  ySp[t] = sigm(o1);
  float o2 = 0.f;
  for (int j = 0; j < 4; ++j) {
    float hsum = 0.f;
    for (int k = 0; k < Cc; ++k) hsum += ya[k] * fc_w1[j * Cc + k];
    o2 += fmaxf(hsum, 0.f) * fc_w2[c * 4 + j];
  }
  yCh[t] = sigm(o2);
}

// ---------------------------------------------------------------- kernel 3
__global__ __launch_bounds__(256) void cbam_offconv(const float* __restrict__ x,
                                                    const float* __restrict__ off_w,
                                                    const float* __restrict__ off_b,
                                                    float* __restrict__ offB,
                                                    float* __restrict__ maskB) {
  const int idx = blockIdx.x * 256 + threadIdx.x;
  if (idx >= Bq * 27 * HW) return;
  const int p = idx % HW;
  const int t = idx / HW;
  const int oc = t % 27, b = t / 27;
  const int ph = p / Ww, pw = p % Ww;
  const float* xb = x + (size_t)b * Cc * HW;
  const float* wq = off_w + oc * Cc * KK9;
  float acc = off_b[oc];
  for (int ic = 0; ic < Cc; ++ic) {
    const float* xc = xb + ic * HW;
    const float* wk = wq + ic * KK9;
#pragma unroll
    for (int kh = 0; kh < 3; ++kh) {
      const int hy = ph + kh - 1;
      if (hy < 0 || hy >= Hh) continue;
#pragma unroll
      for (int kw = 0; kw < 3; ++kw) {
        const int wx = pw + kw - 1;
        if (wx < 0 || wx >= Ww) continue;
        acc += xc[hy * Ww + wx] * wk[kh * 3 + kw];
      }
    }
  }
  if (oc < 18) offB[(size_t)(b * 18 + oc) * HW + p] = acc;
  else         maskB[(size_t)(b * 9 + (oc - 18)) * HW + p] = sigm(acc);
}

// ---------------------------------------------------------------- kernel 4
// DCNv2: bilinear gather (VALU) + 32x288 contraction on WMMA.
// Weights f16, pre-permuted, DMA'd into LDS by the Tensor Data Mover.
__global__ __launch_bounds__(256) void cbam_dcn(const float* __restrict__ x,
                                                const float* __restrict__ offB,
                                                const float* __restrict__ maskB,
                                                const _Float16* __restrict__ wTab,
                                                const float* __restrict__ dcn_b,
                                                float* __restrict__ x3) {
  __shared__ _Float16 wLds[KK9 * Cc * Cc];       // [k][m][ic]
  if (threadIdx.x < 32) {
    tdm_load_f16(&wLds[0], wTab, KK9 * Cc * Cc);
    __builtin_amdgcn_s_wait_tensorcnt(0);
  }
  __syncthreads();

  const int wave = threadIdx.x >> 5, lane = threadIdx.x & 31;
  const int n = lane & 15, hf = lane >> 4;
  const int tile = blockIdx.x * 8 + wave;
  const int pix0 = tile * 16;
  const int b = pix0 / HW, p0 = pix0 % HW;
  const int p = p0 + n;
  const int ph = p / Ww, pw = p % Ww;
  const float* xb = x + (size_t)b * Cc * HW;
  const float* ob = offB + (size_t)b * 18 * HW;
  const float* mb = maskB + (size_t)b * 9 * HW;

  v8f acc0 = {}; v8f acc1 = {};
  for (int k = 0; k < KK9; ++k) {
    const float offy = ob[(k * 2 + 0) * HW + p];
    const float offx = ob[(k * 2 + 1) * HW + p];
    const float msk = mb[k * HW + p];
    const float py = (float)ph + (float)(k / 3 - 1) + offy;
    const float px = (float)pw + (float)(k % 3 - 1) + offx;
    const float y0f = floorf(py), x0f = floorf(px);
    const float wy = py - y0f, wx = px - x0f;
    const int y0 = (int)y0f, x0 = (int)x0f;
    const int y1 = y0 + 1, x1 = x0 + 1;
    const bool vy0 = (y0 >= 0) && (y0 <= Hh - 1);
    const bool vy1 = (y1 >= 0) && (y1 <= Hh - 1);
    const bool vx0 = (x0 >= 0) && (x0 <= Ww - 1);
    const bool vx1 = (x1 >= 0) && (x1 <= Ww - 1);
    const int yc0 = y0 < 0 ? 0 : (y0 > Hh - 1 ? Hh - 1 : y0);
    const int yc1 = y1 < 0 ? 0 : (y1 > Hh - 1 ? Hh - 1 : y1);
    const int xc0 = x0 < 0 ? 0 : (x0 > Ww - 1 ? Ww - 1 : x0);
    const int xc1 = x1 < 0 ? 0 : (x1 > Ww - 1 ? Ww - 1 : x1);
    const float w00 = (vy0 && vx0) ? (1.f - wy) * (1.f - wx) * msk : 0.f;
    const float w01 = (vy0 && vx1) ? (1.f - wy) * wx * msk : 0.f;
    const float w10 = (vy1 && vx0) ? wy * (1.f - wx) * msk : 0.f;
    const float w11 = (vy1 && vx1) ? wy * wx * msk : 0.f;
    const int i00 = yc0 * Ww + xc0, i01 = yc0 * Ww + xc1;
    const int i10 = yc1 * Ww + xc0, i11 = yc1 * Ww + xc1;

    Frag16 bf, a0, a1;
#pragma unroll
    for (int hh = 0; hh < 16; ++hh) {
      const int ic = hf * 16 + hh;                       // B: K rows of column n
      const float* xc = xb + ic * HW;
      const float sv = xc[i00] * w00 + xc[i01] * w01 + xc[i10] * w10 + xc[i11] * w11;
      bf.e[hh] = (_Float16)sv;
      const int icA = hf * 16 + hh;   // A: m=lane&15
      a0.e[hh] = wLds[(k * Cc + n) * Cc + icA];
      a1.e[hh] = wLds[(k * Cc + 16 + n) * Cc + icA];
    }
    acc0 = __builtin_amdgcn_wmma_f32_16x16x32_f16(false, a0.vec, false, bf.vec,
                                                  (short)0, acc0, false, false);
    acc1 = __builtin_amdgcn_wmma_f32_16x16x32_f16(false, a1.vec, false, bf.vec,
                                                  (short)0, acc1, false, false);
  }
  float* x3b = x3 + (size_t)b * Cc * HW;
#pragma unroll
  for (int r = 0; r < 8; ++r) {
    const int m0 = r + hf * 8;
    x3b[m0 * HW + p] = acc0[r] + dcn_b[m0];
    const int m1 = m0 + 16;
    x3b[m1 * HW + p] = acc1[r] + dcn_b[m1];
  }
}

// ---------------------------------------------------------------- kernel 5
__global__ __launch_bounds__(256) void cbam_switch(const float* __restrict__ x,
                                                   const float* __restrict__ x3,
                                                   const _Float16* __restrict__ w1Tab,
                                                   const float* __restrict__ sw_b1,
                                                   const float* __restrict__ sw_w2,
                                                   const float* __restrict__ sw_b2,
                                                   const float* __restrict__ ySp,
                                                   const float* __restrict__ yCh,
                                                   float* __restrict__ xo) {
  __shared__ _Float16 w1Lds[Cc * 64];
  __shared__ _Float16 fTile[8][64 * 16];
  __shared__ _Float16 h1Tile[8][Cc * 16];
  __shared__ float swTile[8][2][16];
  if (threadIdx.x < 32) {
    tdm_load_f16(&w1Lds[0], w1Tab, Cc * 64);
    __builtin_amdgcn_s_wait_tensorcnt(0);
  }
  __syncthreads();

  const int wave = threadIdx.x >> 5, lane = threadIdx.x & 31;
  const int n = lane & 15, hf = lane >> 4;
  const int tile = blockIdx.x * 8 + wave;
  const int pix0 = tile * 16;
  const int b = pix0 / HW, p0 = pix0 % HW;
  const int p = p0 + n;
  const float* xb = x + (size_t)b * Cc * HW;
  const float* x3b = x3 + (size_t)b * Cc * HW;
  _Float16* fT = fTile[wave];
  for (int ii = 0; ii < 32; ++ii) {
    const int feat = hf * 32 + ii;
    const float v = (feat < 32) ? xb[feat * HW + p] : x3b[(feat - 32) * HW + p];
    fT[feat * 16 + n] = (_Float16)v;
  }
  __syncthreads();

  v8f acc0 = {}; v8f acc1 = {};
#pragma unroll
  for (int kb = 0; kb < 2; ++kb) {
    Frag16 a0, a1, bf;
#pragma unroll
    for (int hh = 0; hh < 16; ++hh) {
      const int kk = kb * 32 + hf * 16 + hh;
      a0.e[hh] = w1Lds[n * 64 + kk];
      a1.e[hh] = w1Lds[(16 + n) * 64 + kk];
      const int kkb = kb * 32 + hf * 16 + hh;
      bf.e[hh] = fT[kkb * 16 + n];
    }
    acc0 = __builtin_amdgcn_wmma_f32_16x16x32_f16(false, a0.vec, false, bf.vec,
                                                  (short)0, acc0, false, false);
    acc1 = __builtin_amdgcn_wmma_f32_16x16x32_f16(false, a1.vec, false, bf.vec,
                                                  (short)0, acc1, false, false);
  }
  _Float16* h1T = h1Tile[wave];
#pragma unroll
  for (int r = 0; r < 8; ++r) {
    const int m0 = r + hf * 8;
    h1T[m0 * 16 + n] = (_Float16)fmaxf(acc0[r] + sw_b1[m0], 0.f);
    const int m1 = m0 + 16;
    h1T[m1 * 16 + n] = (_Float16)fmaxf(acc1[r] + sw_b1[m1], 0.f);
  }
  __syncthreads();
  if (hf == 0) {
    float s0 = sw_b2[0], s1 = sw_b2[1];
    for (int c = 0; c < Cc; ++c) {
      const float hv = (float)h1T[c * 16 + n];
      s0 += hv * sw_w2[c];
      s1 += hv * sw_w2[Cc + c];
    }
    swTile[wave][0][n] = sigm(s0);
    swTile[wave][1][n] = sigm(s1);
  }
  __syncthreads();
  const float s0 = swTile[wave][0][n], s1 = swTile[wave][1][n];
  float* xob = xo + (size_t)b * Cc * HW;
  for (int ii = 0; ii < 16; ++ii) {
    const int c = hf * 16 + ii;
    const float xv = xb[c * HW + p];
    xob[c * HW + p] = xv + ySp[b * Cc + c] * s0 + yCh[b * Cc + c] * s1;
  }
}

// ---------------------------------------------------------------- kernel 6
// 4-head memory attention; B fragments are single v16h loads from the
// precomputed tables (L2-resident), A fragment shared across heads.
#define ATT_WAVES 2
__global__ __launch_bounds__(32 * ATT_WAVES) void cbam_attn(const float* __restrict__ xo,
                                                            const _Float16* __restrict__ scoreB,
                                                            const _Float16* __restrict__ recB,
                                                            float* __restrict__ out) {
  __shared__ _Float16 scores[ATT_WAVES][MS * 16];
  __shared__ float red[ATT_WAVES][2][16];
  const int wave = threadIdx.x >> 5, lane = threadIdx.x & 31;
  const int n = lane & 15, hf = lane >> 4;
  const int tile = blockIdx.x * ATT_WAVES + wave;
  const int pix0 = tile * 16;
  const int b = pix0 / HW, p0 = pix0 % HW;
  const int p = p0 + n;
  const float* xob = xo + (size_t)b * Cc * HW;
  float* outb = out + (size_t)b * Cc * HW;
  _Float16* sc = scores[wave];
  const float scale = 0.35355339059327373f;            // 1/sqrt(8)

  Frag16 qa;
#pragma unroll
  for (int hh = 0; hh < 16; ++hh) {
    const int c = hf * 16 + hh;
    qa.e[hh] = (_Float16)xob[c * HW + p];
  }

  for (int hd = 0; hd < MH; ++hd) {
    // scores: one WMMA per 16x16 tile (K=8 padded to 32 via zeroed rows)
    for (int mt = 0; mt < MS / 16; ++mt) {
      Frag16 bf;
      bf.vec = *(const v16h*)(scoreB + ((size_t)((hd * 32 + mt) * 32 + lane) << 4));
      v8f acc = {};
      acc = __builtin_amdgcn_wmma_f32_16x16x32_f16(false, qa.vec, false, bf.vec,
                                                   (short)0, acc, false, false);
#pragma unroll
      for (int r = 0; r < 8; ++r) {
        const int m = mt * 16 + r + hf * 8;
        sc[m * 16 + n] = (_Float16)(acc[r] * scale);
      }
    }
    __syncthreads();

    // softmax over 512 per pixel column; 2 lanes per pixel
    const int mBase = hf * (MS / 2);
    float mx = -1e30f;
    for (int m = 0; m < MS / 2; ++m) mx = fmaxf(mx, (float)sc[(mBase + m) * 16 + n]);
    red[wave][hf][n] = mx;
    __syncthreads();
    mx = fmaxf(mx, red[wave][1 - hf][n]);
    float sum = 0.f;
    for (int m = 0; m < MS / 2; ++m) sum += __expf((float)sc[(mBase + m) * 16 + n] - mx);
    __syncthreads();
    red[wave][hf][n] = sum;
    __syncthreads();
    sum += red[wave][1 - hf][n];
    const float inv = 1.f / sum;
    for (int m = 0; m < MS / 2; ++m) {
      const float s = (float)sc[(mBase + m) * 16 + n];
      sc[(mBase + m) * 16 + n] = (_Float16)(__expf(s - mx) * inv);
    }
    __syncthreads();

    // rec = att(16x512) @ mem(512x8), K accumulated over 16 WMMA tiles
    v8f racc = {};
    for (int mt = 0; mt < MS / 32; ++mt) {
      Frag16 af, bf;
#pragma unroll
      for (int hh = 0; hh < 16; ++hh) {
        const int kkA = hf * 16 + hh;
        af.e[hh] = sc[(mt * 32 + kkA) * 16 + n];
      }
      bf.vec = *(const v16h*)(recB + ((size_t)((hd * 16 + mt) * 32 + lane) << 4));
      racc = __builtin_amdgcn_wmma_f32_16x16x32_f16(false, af.vec, false, bf.vec,
                                                    (short)0, racc, false, false);
    }
    __syncthreads();

    if (n < HD) {
      const int c = hd * HD + n;
#pragma unroll
      for (int r = 0; r < 8; ++r) {
        const int pix = r + hf * 8;
        const int pp = p0 + pix;
        outb[c * HW + pp] = xob[c * HW + pp] + racc[r];
      }
    }
  }
}

// ---------------------------------------------------------------- launch
extern "C" void kernel_launch(void* const* d_in, const int* in_sizes, int n_in,
                              void* d_out, int out_size, void* d_ws, size_t ws_size,
                              hipStream_t stream) {
  const float* x     = (const float*)d_in[0];
  const float* fs_w1 = (const float*)d_in[1];
  const float* fs_w2 = (const float*)d_in[2];
  const float* fc_w1 = (const float*)d_in[3];
  const float* fc_w2 = (const float*)d_in[4];
  const float* sw_w1 = (const float*)d_in[5];
  const float* sw_b1 = (const float*)d_in[6];
  const float* sw_w2 = (const float*)d_in[7];
  const float* sw_b2 = (const float*)d_in[8];
  const float* off_w = (const float*)d_in[9];
  const float* off_b = (const float*)d_in[10];
  const float* dcn_w = (const float*)d_in[11];
  const float* dcn_b = (const float*)d_in[12];
  const float* memw  = (const float*)d_in[13];
  float* out = (float*)d_out;

  float* ws = (float*)d_ws;
  float* yAvg  = ws;                                   // 128
  float* ySp   = ws + 128;                             // 128
  float* yCh   = ws + 256;                             // 128
  float* offB  = ws + 384;                             // 4*18*HW
  float* maskB = offB + (size_t)Bq * 18 * HW;          // 4*9*HW
  float* x3    = maskB + (size_t)Bq * 9 * HW;          // 4*32*HW
  float* xoB   = x3 + (size_t)Bq * Cc * HW;            // 4*32*HW
  _Float16* h16    = (_Float16*)(xoB + (size_t)Bq * Cc * HW);  // 32B-aligned
  _Float16* wTab   = h16;                              // 9216
  _Float16* w1Tab  = wTab + KK9 * Cc * Cc;             // 2048
  _Float16* scoreB = w1Tab + Cc * 64;                  // 65536
  _Float16* recB   = scoreB + MH * 32 * 32 * 16;       // 32768

  const int prepN = KK9 * Cc * Cc + Cc * 64 + MH * 32 * 32 * 16 + MH * 16 * 32 * 16;
  cbam_prep<<<(prepN + 255) / 256, 256, 0, stream>>>(dcn_w, sw_w1, memw,
                                                     wTab, w1Tab, scoreB, recB);
  cbam_mean<<<Bq * Cc, 256, 0, stream>>>(x, yAvg);
  cbam_se<<<1, 128, 0, stream>>>(yAvg, fs_w1, fs_w2, fc_w1, fc_w2, ySp, yCh);
  cbam_offconv<<<(Bq * 27 * HW) / 256, 256, 0, stream>>>(x, off_w, off_b, offB, maskB);
  cbam_dcn<<<(Bq * HW) / (16 * 8), 256, 0, stream>>>(x, offB, maskB, wTab, dcn_b, x3);
  cbam_switch<<<(Bq * HW) / (16 * 8), 256, 0, stream>>>(x, x3, w1Tab, sw_b1, sw_w2,
                                                        sw_b2, ySp, yCh, xoB);
  cbam_attn<<<(Bq * HW) / (16 * ATT_WAVES), 32 * ATT_WAVES, 0, stream>>>(xoB, scoreB,
                                                                         recB, out);
}
